// FlashBlockSparseAttention_84963043049654
// MI455X (gfx1250) — matrix operand order repaired and checked by reading, not compiled, not run on hardware
//
#include <hip/hip_runtime.h>

typedef __bf16 bf16_t;
typedef __attribute__((ext_vector_type(16))) __bf16 v16bf;
typedef __attribute__((ext_vector_type(8)))  __bf16 v8bf;
typedef __attribute__((ext_vector_type(8)))  float  v8f;

// ---------------------------------------------------------------------------
// CDNA5 async global->LDS copy (ASYNCcnt-tracked, ISA ch.10 / 15.18.3).
// Each EXEC-enabled lane copies 16B: vdst = 32-bit LDS byte offset,
// vaddr = 64-bit global address.  Generic->LDS offset: low 32 bits.
// ---------------------------------------------------------------------------
__device__ __forceinline__ void async_copy_b128(const bf16_t* gsrc, bf16_t* ldst) {
    unsigned loff = (unsigned)(unsigned long long)ldst;
    asm volatile("global_load_async_to_lds_b128 %0, %1, off"
                 :: "v"(loff), "v"(gsrc) : "memory");
}
__device__ __forceinline__ void wait_async() {
#if __has_builtin(__builtin_amdgcn_s_wait_asynccnt)
    __builtin_amdgcn_s_wait_asynccnt(0);
#else
    asm volatile("s_wait_asynccnt 0" ::: "memory");
#endif
}

// ---------------------------------------------------------------------------
// Fragment loaders per CDNA5 ISA 7.12.2 (wave32, 16-bit operands).
// A 16x32 (MxK): lane m=l%16, kb=(l>=16)?8:0; two 16B chunks at +0,+16 elems.
// B 32x16 (KxN) from an [n][k]-major LDS tile: lane n=l%16, kb=(l>=16)?8:0;
//   two 16B chunks at +0,+16 elems (same K order as A).
// ---------------------------------------------------------------------------
__device__ __forceinline__ v16bf load_a_frag(const bf16_t* base, int ld, int lane) {
    int m  = lane & 15;
    int kb = (lane >> 4) << 3;
    const bf16_t* p = base + m * ld + kb;
    union { v16bf v; v8bf h[2]; } u;
    u.h[0] = *(const v8bf*)(p);
    u.h[1] = *(const v8bf*)(p + 16);
    return u.v;
}
__device__ __forceinline__ v16bf load_b_frag(const bf16_t* base, int ld, int lane) {
    int n  = lane & 15;
    int kb = (lane >> 4) << 3;
    const bf16_t* p = base + n * ld + kb;
    union { v16bf v; v8bf h[2]; } u;
    u.h[0] = *(const v8bf*)(p);
    u.h[1] = *(const v8bf*)(p + 16);
    return u.v;
}

// f32 -> bf16 tile staging helpers (conversion requires the VGPR path).
__device__ __forceinline__ void stage_x_tile(const float* __restrict__ X, int m0,
                                             int kt, int tid, bf16_t* As) {
    for (int i = 0; i < 4; ++i) {
        int s = i * 256 + tid;               // 1024 float4 slots, coalesced
        int r = s >> 3, kq = (s & 7) << 2;
        float4 f = *(const float4*)(X + (size_t)(m0 + r) * 1024 + kt + kq);
        bf16_t* d = As + r * 40 + kq;
        d[0] = (bf16_t)f.x; d[1] = (bf16_t)f.y;
        d[2] = (bf16_t)f.z; d[3] = (bf16_t)f.w;
    }
}
__device__ __forceinline__ void stage_w_tile(const float* __restrict__ W, int N,
                                             int n0, int kt, int tid, bf16_t* Bs) {
    for (int i = 0; i < 4; ++i) {
        int s  = i * 256 + tid;
        int kk = s >> 5, nq = (s & 31) << 2;
        float4 f = *(const float4*)(W + (size_t)(kt + kk) * N + n0 + nq);
        Bs[(nq + 0) * 40 + kk] = (bf16_t)f.x;
        Bs[(nq + 1) * 40 + kk] = (bf16_t)f.y;
        Bs[(nq + 2) * 40 + kk] = (bf16_t)f.z;
        Bs[(nq + 3) * 40 + kk] = (bf16_t)f.w;
    }
}

// ---------------------------------------------------------------------------
// Kernel 1: QKV projection, double-buffered LDS.
// C[4096,3072] = bf16(X) @ bf16(Wqkv) scattered to q/k/v [B,H,S,64] bf16.
// ---------------------------------------------------------------------------
__global__ __launch_bounds__(256) void qkv_gemm_kernel(
    const float* __restrict__ X, const float* __restrict__ W,
    bf16_t* __restrict__ qB, bf16_t* __restrict__ kB, bf16_t* __restrict__ vB)
{
    const int K = 1024, N = 3072, LD = 40;
    alignas(16) __shared__ bf16_t As[2][128 * LD];
    alignas(16) __shared__ bf16_t Bs[2][128 * LD];   // W tile stored [n][k]

    const int tid  = threadIdx.x;
    const int lane = tid & 31, wave = tid >> 5;
    const int m0 = blockIdx.y * 128;
    const int n0 = blockIdx.x * 128;
    const int wr = (wave & 3) * 32;
    const int wc = (wave >> 2) * 64;

    v8f acc[2][4] = {};

    stage_x_tile(X, m0, 0, tid, As[0]);
    stage_w_tile(W, N, n0, 0, tid, Bs[0]);
    __syncthreads();

    for (int kt = 0; kt < K; kt += 32) {
        int cur = (kt >> 5) & 1;
        if (kt + 32 < K) {                   // stage next tile during compute
            __builtin_prefetch(X + (size_t)(m0 + (tid >> 1)) * K + kt + 32, 0, 3);
            stage_x_tile(X, m0, kt + 32, tid, As[cur ^ 1]);
            stage_w_tile(W, N, n0, kt + 32, tid, Bs[cur ^ 1]);
        }
        v16bf a[2], b[4];
        for (int r = 0; r < 2; ++r) a[r] = load_a_frag(As[cur] + (wr + r * 16) * LD, LD, lane);
        for (int c = 0; c < 4; ++c) b[c] = load_b_frag(Bs[cur] + (wc + c * 16) * LD, LD, lane);
        for (int r = 0; r < 2; ++r)
            for (int c = 0; c < 4; ++c)
                acc[r][c] = __builtin_amdgcn_wmma_f32_16x16x32_bf16(
                    false, a[r], false, b[c], (short)0, acc[r][c], false, false);
        __syncthreads();
    }

    const int half = lane >> 4, nloc = lane & 15;
    for (int r = 0; r < 2; ++r)
        for (int c = 0; c < 4; ++c) {
            int gn = n0 + wc + c * 16 + nloc;
            int which = gn >> 10;            // 0=q 1=k 2=v
            int d     = gn & 1023;
            int h     = d >> 6, di = d & 63;
            bf16_t* dst = (which == 0) ? qB : (which == 1) ? kB : vB;
            for (int j = 0; j < 8; ++j) {
                int row = m0 + wr + r * 16 + j + (half << 3);
                int b_ = row >> 11, s_ = row & 2047;
                dst[(((size_t)(b_ * 16 + h)) * 2048 + s_) * 64 + di] =
                    (bf16_t)acc[r][c][j];
            }
        }
}

// ---------------------------------------------------------------------------
// Kernel 2: causal flash attention, double-buffered K/V tiles; K tile staged
// with async global->LDS (ASYNCcnt), V transposed on store.
// ---------------------------------------------------------------------------
__global__ __launch_bounds__(128) void flash_attn_kernel(
    const bf16_t* __restrict__ qB, const bf16_t* __restrict__ kB,
    const bf16_t* __restrict__ vB, bf16_t* __restrict__ aO)
{
    const int S = 2048, LD = 72;
    const int qt = blockIdx.x, h = blockIdx.y, b = blockIdx.z;
    const int bh = b * 16 + h;
    const bf16_t* Q  = qB + (size_t)bh * S * 64;
    const bf16_t* Kp = kB + (size_t)bh * S * 64;
    const bf16_t* Vp = vB + (size_t)bh * S * 64;

    const int tid = threadIdx.x, lane = tid & 31, wave = tid >> 5;
    const int q0 = qt * 64, qrow0 = q0 + wave * 16;

    alignas(16) __shared__ bf16_t Ks[2][64 * LD];    // [key][d]
    alignas(16) __shared__ bf16_t Vs[2][64 * LD];    // [d][key] (transposed)
    alignas(16) __shared__ bf16_t Ps[4][16 * LD];    // per-wave P staging

    auto stage_kv = [&](int kt2, int buf) {
        int k0 = kt2 * 64;
        for (int i = 0; i < 4; ++i) {
            int s8 = i * 128 + tid;          // 512 8-elem slots
            int r = s8 >> 3, c8 = (s8 & 7) << 3;
            // K: async DMA straight into LDS (already bf16, target layout)
            async_copy_b128(Kp + (size_t)(k0 + r) * 64 + c8, Ks[buf] + r * LD + c8);
            // V: needs transpose -> register path
            v8bf vv = *(const v8bf*)(Vp + (size_t)(k0 + r) * 64 + c8);
            for (int e = 0; e < 8; ++e) Vs[buf][(c8 + e) * LD + r] = vv[e];
        }
    };

    // Q A-fragments straight from global ([B,H,S,64] bf16, d contiguous).
    v16bf qa[2];
    for (int s = 0; s < 2; ++s) {
        int m = lane & 15, kb = (lane >> 4) << 3;
        const bf16_t* p = Q + (size_t)(qrow0 + m) * 64 + s * 32 + kb;
        union { v16bf v; v8bf hh[2]; } u;
        u.hh[0] = *(const v8bf*)(p);
        u.hh[1] = *(const v8bf*)(p + 16);
        qa[s] = u.v;
    }

    v8f o[4] = {};
    float mrow[8], lrow[8];
    for (int j = 0; j < 8; ++j) { mrow[j] = -1e30f; lrow[j] = 0.f; }
    const float scale = 0.03125f;            // 1/sqrt(D) = 1/32
    const int half = lane >> 4, nloc = lane & 15;

    stage_kv(0, 0);
    wait_async();
    __syncthreads();

    for (int kt = 0; kt <= qt; ++kt) {       // causal: skip fully-masked tiles
        int cur = kt & 1;
        int k0 = kt * 64;
        if (kt + 1 <= qt) stage_kv(kt + 1, cur ^ 1);   // overlap with compute

        // S = Q @ K^T  (4 key-subtiles x 2 d-steps)
        v8f sc[4] = {};
        for (int c = 0; c < 4; ++c)
            for (int s = 0; s < 2; ++s) {
                v16bf bk = load_b_frag(Ks[cur] + (c * 16) * LD + s * 32, LD, lane);
                sc[c] = __builtin_amdgcn_wmma_f32_16x16x32_bf16(
                    false, qa[s], false, bk, (short)0, sc[c], false, false);
            }

        // scale + causal mask + row max (C layout: lane=col, vgpr j=row)
        float rmax[8];
        for (int j = 0; j < 8; ++j) {
            int qi = qrow0 + j + (half << 3);
            float mx = -1e30f;
            for (int c = 0; c < 4; ++c) {
                int ki = k0 + c * 16 + nloc;
                float sv = sc[c][j] * scale;
                sv = (ki <= qi) ? sv : -1e30f;
                sc[c][j] = sv;
                mx = fmaxf(mx, sv);
            }
            rmax[j] = mx;
        }
        for (int off = 1; off < 16; off <<= 1)
            for (int j = 0; j < 8; ++j)
                rmax[j] = fmaxf(rmax[j], __shfl_xor(rmax[j], off, 32));

        float rs[8], alpha[8];
        for (int j = 0; j < 8; ++j) {
            float mnew = fmaxf(mrow[j], rmax[j]);
            alpha[j] = __expf(mrow[j] - mnew);
            mrow[j] = mnew;
            float ssum = 0.f;
            for (int c = 0; c < 4; ++c) {
                float p = __expf(sc[c][j] - mnew);
                sc[c][j] = p;
                ssum += p;
            }
            rs[j] = ssum;
        }
        for (int off = 1; off < 16; off <<= 1)
            for (int j = 0; j < 8; ++j)
                rs[j] += __shfl_xor(rs[j], off, 32);
        for (int j = 0; j < 8; ++j) lrow[j] = lrow[j] * alpha[j] + rs[j];
        for (int c = 0; c < 4; ++c)
            for (int j = 0; j < 8; ++j) o[c][j] *= alpha[j];

        // P: C-layout f32 -> LDS bf16 (per-wave region, same-wave DS in order)
        bf16_t* P = Ps[wave];
        for (int c = 0; c < 4; ++c)
            for (int j = 0; j < 8; ++j)
                P[(j + (half << 3)) * LD + c * 16 + nloc] = (bf16_t)sc[c][j];

        // O += P @ V   (2 key-steps x 4 d-subtiles)
        for (int s = 0; s < 2; ++s) {
            v16bf pa = load_a_frag(P + s * 32, LD, lane);
            for (int c = 0; c < 4; ++c) {
                v16bf vb = load_b_frag(Vs[cur] + (c * 16) * LD + s * 32, LD, lane);
                o[c] = __builtin_amdgcn_wmma_f32_16x16x32_bf16(
                    false, pa, false, vb, (short)0, o[c], false, false);
            }
        }
        wait_async();                        // next-tile K DMA landed
        __syncthreads();
    }

    for (int c = 0; c < 4; ++c)
        for (int j = 0; j < 8; ++j) {
            int row = qrow0 + j + (half << 3);
            int d   = c * 16 + nloc;
            aO[((size_t)b * S + row) * 1024 + h * 64 + d] =
                (bf16_t)(o[c][j] / lrow[j]);
        }
}

// ---------------------------------------------------------------------------
// Kernel 3: output projection + bias, double-buffered; A tile (already bf16)
// staged via async global->LDS DMA.
// ---------------------------------------------------------------------------
__global__ __launch_bounds__(256) void out_gemm_kernel(
    const bf16_t* __restrict__ A, const float* __restrict__ W,
    const float* __restrict__ bias, float* __restrict__ out)
{
    const int K = 1024, N = 1024, LD = 40;
    alignas(16) __shared__ bf16_t As[2][128 * LD];
    alignas(16) __shared__ bf16_t Bs[2][128 * LD];

    const int tid  = threadIdx.x;
    const int lane = tid & 31, wave = tid >> 5;
    const int m0 = blockIdx.y * 128;
    const int n0 = blockIdx.x * 128;
    const int wr = (wave & 3) * 32;
    const int wc = (wave >> 2) * 64;

    auto stage_a = [&](int kt2, int buf) {
        for (int i = 0; i < 2; ++i) {
            int s = i * 256 + tid;           // 512 v8bf slots
            int r = s >> 2, c8 = (s & 3) << 3;
            async_copy_b128(A + (size_t)(m0 + r) * K + kt2 + c8,
                            As[buf] + r * LD + c8);
        }
    };

    v8f acc[2][4] = {};

    stage_a(0, 0);
    stage_w_tile(W, N, n0, 0, tid, Bs[0]);
    wait_async();
    __syncthreads();

    for (int kt = 0; kt < K; kt += 32) {
        int cur = (kt >> 5) & 1;
        if (kt + 32 < K) {
            stage_a(kt + 32, cur ^ 1);       // async DMA overlaps WMMA below
            stage_w_tile(W, N, n0, kt + 32, tid, Bs[cur ^ 1]);
        }
        v16bf a[2], b[4];
        for (int r = 0; r < 2; ++r) a[r] = load_a_frag(As[cur] + (wr + r * 16) * LD, LD, lane);
        for (int c = 0; c < 4; ++c) b[c] = load_b_frag(Bs[cur] + (wc + c * 16) * LD, LD, lane);
        for (int r = 0; r < 2; ++r)
            for (int c = 0; c < 4; ++c)
                acc[r][c] = __builtin_amdgcn_wmma_f32_16x16x32_bf16(
                    false, a[r], false, b[c], (short)0, acc[r][c], false, false);
        wait_async();
        __syncthreads();
    }

    const int half = lane >> 4, nloc = lane & 15;
    for (int r = 0; r < 2; ++r)
        for (int c = 0; c < 4; ++c) {
            int col = n0 + wc + c * 16 + nloc;
            float bval = bias[col];
            for (int j = 0; j < 8; ++j) {
                int row = m0 + wr + r * 16 + j + (half << 3);
                out[(size_t)row * N + col] = acc[r][c][j] + bval;
            }
        }
}

// ---------------------------------------------------------------------------
extern "C" void kernel_launch(void* const* d_in, const int* in_sizes, int n_in,
                              void* d_out, int out_size, void* d_ws, size_t ws_size,
                              hipStream_t stream) {
    const float* x    = (const float*)d_in[0];   // [2,2048,1024]
    const float* Wqkv = (const float*)d_in[1];   // [1024,3072]
    const float* Wout = (const float*)d_in[2];   // [1024,1024]
    const float* bout = (const float*)d_in[3];   // [1024]
    (void)in_sizes; (void)n_in; (void)out_size; (void)ws_size;

    const size_t BUF = (size_t)2 * 16 * 2048 * 64 * sizeof(bf16_t); // 8 MB each
    char* wsb = (char*)d_ws;
    bf16_t* qB = (bf16_t*)(wsb + 0 * BUF);
    bf16_t* kB = (bf16_t*)(wsb + 1 * BUF);
    bf16_t* vB = (bf16_t*)(wsb + 2 * BUF);
    bf16_t* aO = (bf16_t*)(wsb + 3 * BUF);       // [B,S,D] bf16

    qkv_gemm_kernel<<<dim3(3072 / 128, 4096 / 128), 256, 0, stream>>>(
        x, Wqkv, qB, kB, vB);
    flash_attn_kernel<<<dim3(32, 16, 2), 128, 0, stream>>>(qB, kB, vB, aO);
    out_gemm_kernel<<<dim3(1024 / 128, 4096 / 128), 256, 0, stream>>>(
        aO, Wout, bout, (float*)d_out);
}
